// EnhancedMultiHeadAttention_83880711291029
// MI455X (gfx1250) — matrix operand order repaired and checked by reading, not compiled, not run on hardware
//
#include <hip/hip_runtime.h>

typedef _Float16 v16h __attribute__((ext_vector_type(16)));
typedef _Float16 v8h  __attribute__((ext_vector_type(8)));
typedef float    v8f  __attribute__((ext_vector_type(8)));

#define Bsz 4
#define Ssz 1024
#define Dsz 1024
#define Hn  16
#define DK  64
#define NREL 257   // 2*128 + 1

// ---------------- f32 -> f16 conversion of activations ----------------
__global__ __launch_bounds__(256) void cvt_f16_kernel(const float* __restrict__ src,
                                                      _Float16* __restrict__ dst, int n) {
  int i = blockIdx.x * 256 + threadIdx.x;
  if (i < n) dst[i] = (_Float16)src[i];
}

// ---------------- W [n,k] f32  ->  Wt [k,n] f16 (so B-frags load contiguously) ----
__global__ __launch_bounds__(256) void wtrans_f16_kernel(const float* __restrict__ W,
                                                         _Float16* __restrict__ Wt) {
  int i = blockIdx.x * 256 + threadIdx.x;   // i = k*D + n
  int k = i >> 10, n = i & 1023;
  Wt[i] = (_Float16)W[(size_t)n * Dsz + k];
}

// ---------------- WMMA projection: out = X @ W^T + b -------------------------------
// A (16x32 f16): lane row = lane%16, K chunks [h8*8, +8) and [16+h8*8, +8)
// B (32x16 f16): lane col = lane%16, K chunks [h8*8, +8) and [16+h8*8, +8), strided gather from Wt
// C (16x16 f32): vgpr r -> row (lane/16)*8 + r, col = lane%16
__global__ __launch_bounds__(256) void proj_wmma_kernel(
    const _Float16* __restrict__ X,    // [4096, 1024] activations (f16)
    const _Float16* __restrict__ Wt,   // [1024, 1024] Wt[k][n] = W[n][k] (f16)
    const float*    __restrict__ bias, // [1024]
    _Float16* __restrict__ Qout,       // mode 0: [4096, 1024] = [B,S,D]
    _Float16* __restrict__ KTout,      // mode 1: [B, H, 64, S] (transposed K heads)
    int mode) {
  const int wv = threadIdx.x >> 5, ln = threadIdx.x & 31;
  const int l16 = ln & 15, hh = ln >> 4;
  const int tile = blockIdx.x * 8 + wv;        // 16384 tiles total
  const int mt = tile >> 6, nt = tile & 63;
  const int m0 = mt << 4, n0 = nt << 4;

  const _Float16* ap = X + (size_t)(m0 + l16) * Dsz;
  const _Float16* bp = Wt + n0;
  v8f c = {};
  for (int d0 = 0; d0 < Dsz; d0 += 32) {
    v8h alo = *(const v8h*)(ap + d0 + hh * 8);
    v8h ahi = *(const v8h*)(ap + d0 + 16 + hh * 8);
    v16h a;
#pragma unroll
    for (int j = 0; j < 8; ++j) { a[j] = alo[j]; a[j + 8] = ahi[j]; }
    v16h bf;
#pragma unroll
    for (int j = 0; j < 8; ++j) { bf[j] = bp[(size_t)(d0 + hh * 8 + j) * Dsz + l16]; bf[j + 8] = bp[(size_t)(d0 + 16 + hh * 8 + j) * Dsz + l16]; }
    c = __builtin_amdgcn_wmma_f32_16x16x32_f16(false, a, false, bf, (short)0, c, false, false);
  }
  const float bv = bias[n0 + l16];
#pragma unroll
  for (int r = 0; r < 8; ++r) c[r] += bv;

  if (mode == 0) {
#pragma unroll
    for (int r = 0; r < 8; ++r)
      Qout[(size_t)(m0 + hh * 8 + r) * Dsz + n0 + l16] = (_Float16)c[r];
  } else {
    // scatter into KT[b][h][d][s]; per lane the 8 accumulators are 8 consecutive s
    const int dg = n0 + l16, h = dg >> 6, dd = dg & 63;
    const int bb = m0 >> 10, s0 = (m0 & 1023) + hh * 8;
    v8h pk;
#pragma unroll
    for (int r = 0; r < 8; ++r) pk[r] = (_Float16)c[r];
    *(v8h*)(KTout + (((size_t)bb * Hn + h) * DK + dd) * Ssz + s0) = pk;
  }
}

// ---------------- q_mean[b,s,d] = mean_h Q[b,s,h*64+d] ---------------------------
__global__ __launch_bounds__(256) void qmean_kernel(const _Float16* __restrict__ Q,
                                                    float* __restrict__ qmean) {
  int i = blockIdx.x * 256 + threadIdx.x;   // over B*S*64
  int d = i & 63, bs = i >> 6;
  const _Float16* p = Q + (size_t)bs * Dsz + d;
  float acc = 0.f;
#pragma unroll
  for (int h = 0; h < Hn; ++h) acc += (float)p[h * DK];
  qmean[i] = acc * (1.f / 16.f);
}

// ---------------- rel_bias[b,s,p] = scale * dot(q_mean[b,s,:], rel_k[p,:]) -------
__global__ __launch_bounds__(256) void relbias_kernel(const float* __restrict__ qmean,
                                                      const float* __restrict__ relk,
                                                      float* __restrict__ rbias, int total) {
  int i = blockIdx.x * 256 + threadIdx.x;
  if (i >= total) return;
  int p = i % NREL, bs = i / NREL;
  const float* qm = qmean + (size_t)bs * DK;
  const float* rk = relk + (size_t)p * DK;
  float acc = 0.f;
#pragma unroll 8
  for (int j = 0; j < DK; ++j) acc = fmaf(qm[j], rk[j], acc);
  rbias[i] = acc * 0.125f;   // 1/sqrt(64)
}

// ---------------- fused scores + rel-bias + mask + row softmax -------------------
// grid: (S/16, H, B); block 256 (8 waves). Each block owns a 16 x 1024 score tile.
__global__ __launch_bounds__(256) void attn_softmax_kernel(
    const _Float16* __restrict__ Qf,   // [B,S,D] f16
    const _Float16* __restrict__ KT,   // [B,H,64,S] f16
    const float*    __restrict__ rbias,// [B,S,257]
    const int*      __restrict__ mask, // [B,S,S]
    float*          __restrict__ out) {// [B,H,S,S]
  __shared__ float sm[16 * Ssz];       // 64 KB score tile
  const int qt = blockIdx.x, h = blockIdx.y, b = blockIdx.z;
  const int q0 = qt << 4;
  const int wv = threadIdx.x >> 5, ln = threadIdx.x & 31;
  const int l16 = ln & 15, hh = ln >> 4;

  // Q-head A-fragments for both K=32 steps of the d=64 reduction (hoisted)
  const _Float16* ap = Qf + (size_t)(b * Ssz + q0 + l16) * Dsz + h * DK;
  v16h a0, a1;
  {
    v8h lo = *(const v8h*)(ap + hh * 8);
    v8h hi = *(const v8h*)(ap + 16 + hh * 8);
#pragma unroll
    for (int j = 0; j < 8; ++j) { a0[j] = lo[j]; a0[j + 8] = hi[j]; }
    lo = *(const v8h*)(ap + 32 + hh * 8);
    hi = *(const v8h*)(ap + 48 + hh * 8);
#pragma unroll
    for (int j = 0; j < 8; ++j) { a1[j] = lo[j]; a1[j + 8] = hi[j]; }
  }
  const _Float16* kb = KT + ((size_t)b * Hn + h) * DK * Ssz;
  const float* rb = rbias + (size_t)b * Ssz * NREL;
  const int* mbase = mask + (size_t)b * Ssz * Ssz;

  for (int t = 0; t < 8; ++t) {                // 8 column tiles per wave
    const int c0 = ((wv << 3) + t) << 4;
    v8f c = {};
    v16h b0, b1;   // col = c0+l16, d in table order
#pragma unroll
    for (int j = 0; j < 8; ++j) { b0[j] = kb[(size_t)(hh * 8 + j) * Ssz + c0 + l16]; b0[j + 8] = kb[(size_t)(16 + hh * 8 + j) * Ssz + c0 + l16]; }
    c = __builtin_amdgcn_wmma_f32_16x16x32_f16(false, a0, false, b0, (short)0, c, false, false);
#pragma unroll
    for (int j = 0; j < 8; ++j) { b1[j] = kb[(size_t)(32 + hh * 8 + j) * Ssz + c0 + l16]; b1[j + 8] = kb[(size_t)(48 + hh * 8 + j) * Ssz + c0 + l16]; }
    c = __builtin_amdgcn_wmma_f32_16x16x32_f16(false, a1, false, b1, (short)0, c, false, false);

    const int col = c0 + l16;
#pragma unroll
    for (int r = 0; r < 8; ++r) {
      const int row = hh * 8 + r, q = q0 + row;
      int delta = col - q;
      delta = delta < -128 ? -128 : (delta > 128 ? 128 : delta);
      float s = c[r] * 0.125f + rb[(size_t)q * NREL + (delta + 128)];
      if (mbase[(size_t)q * Ssz + col] == 0) s = -1e9f;
      sm[row * Ssz + col] = s;
    }
  }
  __syncthreads();

  // 2 rows per wave; 32 elements per lane; wave32 shfl_xor reductions
#pragma unroll
  for (int i = 0; i < 2; ++i) {
    const int row = wv * 2 + i, q = q0 + row;
    float v[32], m = -3.4028235e38f;
#pragma unroll
    for (int j = 0; j < 32; ++j) { v[j] = sm[row * Ssz + ln + 32 * j]; m = fmaxf(m, v[j]); }
#pragma unroll
    for (int off = 16; off > 0; off >>= 1) m = fmaxf(m, __shfl_xor(m, off, 32));
    float ssum = 0.f;
#pragma unroll
    for (int j = 0; j < 32; ++j) { v[j] = __expf(v[j] - m); ssum += v[j]; }
#pragma unroll
    for (int off = 16; off > 0; off >>= 1) ssum += __shfl_xor(ssum, off, 32);
    const float inv = 1.0f / ssum;
    float* orow = out + ((size_t)(b * Hn + h) * Ssz + q) * Ssz;
#pragma unroll
    for (int j = 0; j < 32; ++j) orow[ln + 32 * j] = v[j] * inv;
  }
}

extern "C" void kernel_launch(void* const* d_in, const int* in_sizes, int n_in,
                              void* d_out, int out_size, void* d_ws, size_t ws_size,
                              hipStream_t stream) {
  (void)in_sizes; (void)n_in; (void)out_size; (void)ws_size;
  const float* query = (const float*)d_in[0];
  const float* key_  = (const float*)d_in[1];
  const int*   mask  = (const int*)d_in[2];
  const float* Wq    = (const float*)d_in[3];
  const float* bq    = (const float*)d_in[4];
  const float* Wk    = (const float*)d_in[5];
  const float* bk    = (const float*)d_in[6];
  const float* relk  = (const float*)d_in[7];
  float* out = (float*)d_out;

  char* p = (char*)d_ws;
  const size_t szAct = (size_t)Bsz * Ssz * Dsz * sizeof(_Float16);  // 8 MB
  const size_t szW   = (size_t)Dsz * Dsz * sizeof(_Float16);        // 2 MB
  _Float16* Aq  = (_Float16*)p; p += szAct;
  _Float16* Ak  = (_Float16*)p; p += szAct;
  _Float16* Wtq = (_Float16*)p; p += szW;
  _Float16* Wtk = (_Float16*)p; p += szW;
  _Float16* Qf  = (_Float16*)p; p += szAct;
  _Float16* KT  = (_Float16*)p; p += szAct;
  float* qmean  = (float*)p;    p += (size_t)Bsz * Ssz * DK * sizeof(float);
  float* rbias  = (float*)p;    p += (size_t)Bsz * Ssz * NREL * sizeof(float);

  const int nAct = Bsz * Ssz * Dsz;                                  // 4M
  cvt_f16_kernel<<<nAct / 256, 256, 0, stream>>>(query, Aq, nAct);
  cvt_f16_kernel<<<nAct / 256, 256, 0, stream>>>(key_, Ak, nAct);
  wtrans_f16_kernel<<<(Dsz * Dsz) / 256, 256, 0, stream>>>(Wq, Wtq);
  wtrans_f16_kernel<<<(Dsz * Dsz) / 256, 256, 0, stream>>>(Wk, Wtk);

  // 16384 output tiles / 8 waves per block = 2048 blocks per projection
  proj_wmma_kernel<<<2048, 256, 0, stream>>>(Aq, Wtq, bq, Qf, (_Float16*)nullptr, 0);
  proj_wmma_kernel<<<2048, 256, 0, stream>>>(Ak, Wtk, bk, (_Float16*)nullptr, KT, 1);

  qmean_kernel<<<(Bsz * Ssz * DK) / 256, 256, 0, stream>>>(Qf, qmean);
  const int nrb = Bsz * Ssz * NREL;
  relbias_kernel<<<(nrb + 255) / 256, 256, 0, stream>>>(qmean, relk, rbias, nrb);

  dim3 g(Ssz / 16, Hn, Bsz);
  attn_softmax_kernel<<<g, 256, 0, stream>>>(Qf, KT, rbias, mask, out);
}
